// GAT_12180527251806
// MI455X (gfx1250) — matrix operand order repaired and checked by reading, not compiled, not run on hardware
//
#include <hip/hip_runtime.h>
#include <hip/hip_bf16.h>

typedef __bf16 bf16_t;
typedef bf16_t bf16x8  __attribute__((ext_vector_type(8)));
typedef bf16_t bf16x16 __attribute__((ext_vector_type(16)));
typedef float  floatx8 __attribute__((ext_vector_type(8)));

__device__ __forceinline__ bf16_t f2bf(float f) {
    unsigned u = __builtin_bit_cast(unsigned, f);
    unsigned r = u + 0x7FFFu + ((u >> 16) & 1u);     // round-to-nearest-even
    unsigned short s = (unsigned short)(r >> 16);
    return __builtin_bit_cast(bf16_t, s);
}
__device__ __forceinline__ float bf2f(bf16_t b) {
    unsigned short s = __builtin_bit_cast(unsigned short, b);
    unsigned u = ((unsigned)s) << 16;
    return __builtin_bit_cast(float, u);
}
__device__ __forceinline__ float lrelu(float x) { return (x >= 0.f) ? x : 0.2f * x; }

// ---------------------------------------------------------------- prep kernels
__global__ void cvt_f32_bf16(const float* __restrict__ in, bf16_t* __restrict__ out, int n) {
    int i = blockIdx.x * 256 + threadIdx.x;
    if (i < n) out[i] = f2bf(in[i]);
}

// in [H][K][N] f32 -> out [H][N][K] bf16 (col-major weights for B-fragments)
__global__ void cvt_w_transpose(const float* __restrict__ in, bf16_t* __restrict__ out,
                                int H, int K, int N) {
    int i = blockIdx.x * 256 + threadIdx.x;
    int total = H * K * N;
    if (i < total) {
        int h = i / (K * N);
        int r = i % (K * N);
        int k = r / N;
        int n = r % N;
        out[(size_t)h * K * N + (size_t)n * K + k] = f2bf(in[i]);
    }
}

__global__ void zero_f32(float* __restrict__ p, int n) {
    int i = blockIdx.x * 256 + threadIdx.x;
    if (i < n) p[i] = 0.f;
}

// ---------------------------------------------------------------- GEMM: C(cm) = A(rm bf16, Mx256) * W(cm bf16, 64x256)
// grid: (16, 1, Z) ; block 128 (4 waves, 16 rows each). C stored col-major [Z][64][1024].
// Register double-buffered: next k-step fragments are loaded before current WMMAs.
__global__ void __launch_bounds__(128)
gat_gemm_bf16(const bf16_t* __restrict__ Abase, const bf16_t* __restrict__ WTbase,
              bf16_t* __restrict__ Cbase, int aStrideB, int wStrideH) {
    const int z    = blockIdx.z;
    const int b    = z & 15;
    const int h    = z >> 4;
    const int lane = threadIdx.x & 31;
    const int wave = threadIdx.x >> 5;
    const int i0   = blockIdx.x * 64 + wave * 16;
    const int m    = lane & 15;
    const int kbA  = (lane < 16) ? 0 : 16;   // A-frag K base (contiguous K order, matches B)
    const int kbB  = (lane < 16) ? 0 : 16;   // B-frag K base

    const bf16_t* A  = Abase + (size_t)b * aStrideB + (size_t)(i0 + m) * 256;
    const bf16_t* WT = WTbase + (size_t)h * wStrideH;

    auto loadA = [&](int k0) {
        bf16x8 alo = *(const bf16x8*)(A + k0 + kbA);
        bf16x8 ahi = *(const bf16x8*)(A + k0 + kbA + 8);
        return __builtin_shufflevector(alo, ahi, 0,1,2,3,4,5,6,7,8,9,10,11,12,13,14,15);
    };
    auto loadB = [&](int k0, int t) {
        return *(const bf16x16*)(WT + k0 + kbB + (size_t)(t * 16 + m) * 256);
    };

    floatx8 acc[4];
#pragma unroll
    for (int t = 0; t < 4; ++t) acc[t] = floatx8{};

    bf16x16 af = loadA(0);
    bf16x16 bb[4];
#pragma unroll
    for (int t = 0; t < 4; ++t) bb[t] = loadB(0, t);

#pragma unroll
    for (int k0 = 0; k0 < 256; k0 += 32) {
        const int k2 = (k0 + 32) & 255;       // last iter reloads k=0 (discarded)
        bf16x16 naf = loadA(k2);
        bf16x16 nbb[4];
#pragma unroll
        for (int t = 0; t < 4; ++t) nbb[t] = loadB(k2, t);
#pragma unroll
        for (int t = 0; t < 4; ++t)
            acc[t] = __builtin_amdgcn_wmma_f32_16x16x32_bf16(
                         false, af, false, bb[t], (short)0, acc[t], false, false);
        af = naf;
#pragma unroll
        for (int t = 0; t < 4; ++t) bb[t] = nbb[t];
    }

    // store col-major: per lane, 8 consecutive rows per accumulator -> one b128 store
    bf16_t* C = Cbase + (size_t)z * (64 * 1024);
    const int roff = (lane < 16) ? 0 : 8;
#pragma unroll
    for (int t = 0; t < 4; ++t) {
        bf16x8 st;
#pragma unroll
        for (int r = 0; r < 8; ++r) st[r] = f2bf(acc[t][r]);
        *(bf16x8*)(C + (size_t)(t * 16 + m) * 1024 + i0 + roff) = st;
    }
}

// ---------------------------------------------------------------- s1/s2 = Wh @ a1, Wh @ a2 (Wh col-major)
__global__ void gat_sdots(const bf16_t* __restrict__ Whcm, const float* __restrict__ abase,
                          float* __restrict__ s1g, float* __restrict__ s2g, int aHeadStride) {
    const int z = blockIdx.x;
    const int h = z >> 4;
    const float* a = abase + (size_t)h * aHeadStride;
    const bf16_t* W = Whcm + (size_t)z * 65536;
    for (int i = threadIdx.x; i < 1024; i += 256) {
        float d1 = 0.f, d2 = 0.f;
#pragma unroll 4
        for (int c = 0; c < 64; ++c) {
            float wv = bf2f(W[(size_t)c * 1024 + i]);   // coalesced across lanes
            d1 += wv * a[c];
            d2 += wv * a[64 + c];
        }
        s1g[z * 1024 + i] = d1;
        s2g[z * 1024 + i] = d2;
    }
}

__global__ void gat_rowmax(const float* __restrict__ s2g, float* __restrict__ s2maxg) {
    __shared__ float red[256];
    const int z = blockIdx.x;
    float mx = -3.4e38f;
    for (int i = threadIdx.x; i < 1024; i += 256) mx = fmaxf(mx, s2g[z * 1024 + i]);
    red[threadIdx.x] = mx;
    __syncthreads();
    for (int s = 128; s > 0; s >>= 1) {
        if (threadIdx.x < s) red[threadIdx.x] = fmaxf(red[threadIdx.x], red[threadIdx.x + s]);
        __syncthreads();
    }
    if (threadIdx.x == 0) s2maxg[z] = red[0];
}

// ---------------------------------------------------------------- fused masked-softmax + (att @ Wh) via WMMA
// grid: (8, 1, Z) ; block 128 (4 waves x 32 rows). B-fragments reused across 2 row-groups
// (8 WMMAs per B-tile load); adj/s2 raw loads for the next j-tile are prefetched into
// registers while the current tile's exp/mask VALU block and WMMAs execute.
// mode 0: write bf16 concat features. mode 1: ELU + mean -> atomics.
__global__ void __launch_bounds__(128)
gat_attn(const float* __restrict__ s1g, const float* __restrict__ s2g,
         const float* __restrict__ s2maxg, const bf16_t* __restrict__ Whcm,
         const int* __restrict__ adj, bf16_t* __restrict__ x2out,
         float* __restrict__ fout, int mode) {
    __shared__ float lpart[4][64];
    __shared__ __align__(16) bf16_t stage[4][32 * 64];

    const int z    = blockIdx.z;
    const int b    = z & 15;
    const int h    = z >> 4;
    const int lane = threadIdx.x & 31;
    const int wave = threadIdx.x >> 5;
    const int i0   = blockIdx.x * 128 + wave * 32;
    const int m    = lane & 15;
    const int kb   = (lane < 16) ? 0 : 16;
    const int kbB  = (lane < 16) ? 0 : 16;
    const int row0 = i0 + m;
    const int row1 = i0 + 16 + m;

    const float s2mx = s2maxg[z];
    const float s1v0 = s1g[z * 1024 + row0];
    const float s1v1 = s1g[z * 1024 + row1];
    const float Mi0  = lrelu(s1v0 + s2mx);   // per-row upper bound on lrelu(s1+s2)
    const float Mi1  = lrelu(s1v1 + s2mx);
    const float* s2p   = s2g + z * 1024;
    const int*   adjp0 = adj + (size_t)b * 1048576 + (size_t)row0 * 1024;
    const int*   adjp1 = adj + (size_t)b * 1048576 + (size_t)row1 * 1024;
    const bf16_t* W    = Whcm + (size_t)z * 65536;

    floatx8 acc[2][4];
#pragma unroll
    for (int g = 0; g < 2; ++g)
#pragma unroll
        for (int t = 0; t < 4; ++t) acc[g][t] = floatx8{};
    float lsum0 = 0.f, lsum1 = 0.f;

    // raw-tile registers (current + next): s2 shared by both row groups
    float4 cs2[4], ns2[4];
    int4   ca0[4], na0[4], ca1[4], na1[4];
    auto loadRaw = [&](int j0, float4* s2v, int4* a0v, int4* a1v) {
#pragma unroll
        for (int q = 0; q < 2; ++q) {
            s2v[q]     = *(const float4*)(s2p + j0 + kb + 4 * q);
            s2v[2 + q] = *(const float4*)(s2p + j0 + kb + 8 + 4 * q);
            a0v[q]     = *(const int4*)(adjp0 + j0 + kb + 4 * q);
            a0v[2 + q] = *(const int4*)(adjp0 + j0 + kb + 8 + 4 * q);
            a1v[q]     = *(const int4*)(adjp1 + j0 + kb + 4 * q);
            a1v[2 + q] = *(const int4*)(adjp1 + j0 + kb + 8 + 4 * q);
        }
    };
    loadRaw(0, cs2, ca0, ca1);

    for (int j0 = 0; j0 < 1024; j0 += 32) {
        // B fragments for this tile: issued first, consumed last (latency hidden by VALU)
        bf16x16 bb[4];
#pragma unroll
        for (int t = 0; t < 4; ++t)
            bb[t] = *(const bf16x16*)(W + j0 + kbB + (size_t)(t * 16 + m) * 1024);

        // prefetch next tile's adj/s2 (overlaps exp block + WMMAs)
        const int j2 = (j0 + 32 < 1024) ? (j0 + 32) : 0;
        loadRaw(j2, ns2, na0, na1);

        const float sv[16] = {cs2[0].x, cs2[0].y, cs2[0].z, cs2[0].w,
                              cs2[1].x, cs2[1].y, cs2[1].z, cs2[1].w,
                              cs2[2].x, cs2[2].y, cs2[2].z, cs2[2].w,
                              cs2[3].x, cs2[3].y, cs2[3].z, cs2[3].w};
        const int  a0[16] = {ca0[0].x, ca0[0].y, ca0[0].z, ca0[0].w,
                             ca0[1].x, ca0[1].y, ca0[1].z, ca0[1].w,
                             ca0[2].x, ca0[2].y, ca0[2].z, ca0[2].w,
                             ca0[3].x, ca0[3].y, ca0[3].z, ca0[3].w};
        const int  a1[16] = {ca1[0].x, ca1[0].y, ca1[0].z, ca1[0].w,
                             ca1[1].x, ca1[1].y, ca1[1].z, ca1[1].w,
                             ca1[2].x, ca1[2].y, ca1[2].z, ca1[2].w,
                             ca1[3].x, ca1[3].y, ca1[3].z, ca1[3].w};

        bf16x16 pf0, pf1;
#pragma unroll
        for (int q = 0; q < 16; ++q) {
            float e0 = lrelu(s1v0 + sv[q]);
            float p0 = (a0[q] > 0) ? __expf(e0 - Mi0) : 0.f;
            lsum0 += p0;
            pf0[q] = f2bf(p0);
            float e1 = lrelu(s1v1 + sv[q]);
            float p1 = (a1[q] > 0) ? __expf(e1 - Mi1) : 0.f;
            lsum1 += p1;
            pf1[q] = f2bf(p1);
        }

#pragma unroll
        for (int t = 0; t < 4; ++t) {
            acc[0][t] = __builtin_amdgcn_wmma_f32_16x16x32_bf16(
                            false, pf0, false, bb[t], (short)0, acc[0][t], false, false);
            acc[1][t] = __builtin_amdgcn_wmma_f32_16x16x32_bf16(
                            false, pf1, false, bb[t], (short)0, acc[1][t], false, false);
        }

#pragma unroll
        for (int q = 0; q < 4; ++q) { cs2[q] = ns2[q]; ca0[q] = na0[q]; ca1[q] = na1[q]; }
    }

    lpart[wave][lane]      = lsum0;
    lpart[wave][32 + lane] = lsum1;
    __syncthreads();

    const int ro = (lane < 16) ? 0 : 8;
    float invl[2][8];
#pragma unroll
    for (int g = 0; g < 2; ++g)
#pragma unroll
        for (int r = 0; r < 8; ++r) {
            float l = lpart[wave][g * 32 + ro + r] + lpart[wave][g * 32 + ro + r + 16];
            invl[g][r] = 1.0f / fmaxf(l, 1e-30f);
        }

    if (mode == 0) {
        // transpose 32x64 C tile through LDS -> vectorized row-major bf16 stores
#pragma unroll
        for (int g = 0; g < 2; ++g)
#pragma unroll
            for (int t = 0; t < 4; ++t)
#pragma unroll
                for (int r = 0; r < 8; ++r)
                    stage[wave][(g * 16 + ro + r) * 64 + t * 16 + m] =
                        f2bf(acc[g][t][r] * invl[g][r]);
        __syncthreads();
        bf16_t* dst = x2out + (size_t)b * (1024 * 256) + (size_t)i0 * 256 + h * 64;
#pragma unroll
        for (int q = 0; q < 8; ++q) {
            int idx = q * 32 + lane;   // 256 chunks of 8 bf16
            int rr  = idx >> 3;        // 0..31
            int seg = idx & 7;         // 0..7
            bf16x8 v = *(const bf16x8*)(&stage[wave][rr * 64 + seg * 8]);
            *(bf16x8*)(dst + (size_t)rr * 256 + seg * 8) = v;
        }
    } else {
        const float scale = 1.0f / 1024.0f;
#pragma unroll
        for (int t = 0; t < 4; ++t) {
            float s = 0.f;
#pragma unroll
            for (int g = 0; g < 2; ++g)
#pragma unroll
                for (int r = 0; r < 8; ++r) {
                    float v = acc[g][t][r] * invl[g][r];
                    v = (v > 0.f) ? v : (__expf(v) - 1.0f);   // ELU
                    s += v;
                }
            atomicAdd(fout + b * 64 + t * 16 + m, s * scale);
        }
    }
}

// ---------------------------------------------------------------- launch
extern "C" void kernel_launch(void* const* d_in, const int* in_sizes, int n_in,
                              void* d_out, int out_size, void* d_ws, size_t ws_size,
                              hipStream_t stream) {
    const float* x       = (const float*)d_in[0];   // (16,1024,256)
    const int*   adj     = (const int*)d_in[1];     // (16,1024,1024)
    const float* W_heads = (const float*)d_in[2];   // (4,256,64)
    const float* a_heads = (const float*)d_in[3];   // (4,128,1)
    const float* W_out   = (const float*)d_in[4];   // (256,64)
    const float* a_out   = (const float*)d_in[5];   // (128,1)
    float* out = (float*)d_out;                     // (16,64)

    char* ws = (char*)d_ws;
    size_t off = 0;
    auto carve = [&](size_t bytes) { char* p = ws + off; off = (off + bytes + 255) & ~(size_t)255; return p; };

    bf16_t* xb     = (bf16_t*)carve((size_t)16 * 1024 * 256 * 2);   // x as bf16 row-major
    bf16_t* WT1    = (bf16_t*)carve((size_t)4 * 64 * 256 * 2);      // W_heads col-major
    bf16_t* WT2    = (bf16_t*)carve((size_t)64 * 256 * 2);          // W_out   col-major
    bf16_t* Whcm1  = (bf16_t*)carve((size_t)64 * 64 * 1024 * 2);    // layer1 Wh, col-major per (h,b)
    float*  s1_1   = (float*)carve((size_t)64 * 1024 * 4);
    float*  s2_1   = (float*)carve((size_t)64 * 1024 * 4);
    float*  s2m_1  = (float*)carve((size_t)64 * 4);
    bf16_t* x2     = (bf16_t*)carve((size_t)16 * 1024 * 256 * 2);   // concat head outputs bf16
    bf16_t* Whcm2  = (bf16_t*)carve((size_t)16 * 64 * 1024 * 2);
    float*  s1_2   = (float*)carve((size_t)16 * 1024 * 4);
    float*  s2_2   = (float*)carve((size_t)16 * 1024 * 4);
    float*  s2m_2  = (float*)carve((size_t)16 * 4);
    (void)ws_size; (void)in_sizes; (void)n_in; (void)out_size;

    // prep
    cvt_f32_bf16<<<(16 * 1024 * 256 + 255) / 256, 256, 0, stream>>>(x, xb, 16 * 1024 * 256);
    cvt_w_transpose<<<(4 * 256 * 64 + 255) / 256, 256, 0, stream>>>(W_heads, WT1, 4, 256, 64);
    cvt_w_transpose<<<(256 * 64 + 255) / 256, 256, 0, stream>>>(W_out, WT2, 1, 256, 64);
    zero_f32<<<4, 256, 0, stream>>>(out, 16 * 64);

    // ---- layer 1 (4 heads x 16 batches = 64 z-slices)
    dim3 gg1(16, 1, 64);
    dim3 ga1(8, 1, 64);
    gat_gemm_bf16<<<gg1, 128, 0, stream>>>(xb, WT1, Whcm1, 1024 * 256, 64 * 256);
    gat_sdots<<<64, 256, 0, stream>>>(Whcm1, a_heads, s1_1, s2_1, 128);
    gat_rowmax<<<64, 256, 0, stream>>>(s2_1, s2m_1);
    gat_attn<<<ga1, 128, 0, stream>>>(s1_1, s2_1, s2m_1, Whcm1, adj, x2, nullptr, 0);

    // ---- layer 2 (16 z-slices)
    dim3 gg2(16, 1, 16);
    dim3 ga2(8, 1, 16);
    gat_gemm_bf16<<<gg2, 128, 0, stream>>>(x2, WT2, Whcm2, 1024 * 256, 0);
    gat_sdots<<<16, 256, 0, stream>>>(Whcm2, a_out, s1_2, s2_2, 0);
    gat_rowmax<<<16, 256, 0, stream>>>(s2_2, s2m_2);
    gat_attn<<<ga2, 128, 0, stream>>>(s1_2, s2_2, s2m_2, Whcm2, adj, nullptr, out, 1);
}
